// SelfModel_30889404792854
// MI455X (gfx1250) — matrix operand order repaired and checked by reading, not compiled, not run on hardware
//
#include <hip/hip_runtime.h>

typedef __attribute__((ext_vector_type(16))) _Float16 v16h;
typedef __attribute__((ext_vector_type(8)))  float    v8f;
typedef __attribute__((ext_vector_type(4)))  float    v4f;

#define BATCH  8192
#define NCOLS  67
#define FACTOR 64

__device__ __forceinline__ float log_sigmoid(float x) {
    // stable: min(x,0) - log1p(exp(-|x|))
    return fminf(x, 0.0f) - log1pf(__expf(-fabsf(x)));
}

__global__ __launch_bounds__(256) void bpr_loss_kernel(
    const int*   __restrict__ one_batch,
    const float* __restrict__ embed_user,
    const float* __restrict__ embed_item,
    float*       __restrict__ ws)
{
    const int lane = threadIdx.x & 31;
    const int wid  = threadIdx.x >> 5;
    const int row  = blockIdx.x * 8 + wid;   // one wave32 per batch row
    const int g    = lane >> 4;              // half-wave group (K-group select)
    const int n    = lane & 15;              // N (item-position within tile)

    const int* brow = one_batch + row * NCOLS;
    const int  uidx = brow[0];
    const float* u  = embed_user + (size_t)uidx * FACTOR;

    // ---- A fragments: user vector, replicated across all 16 M rows ----
    // 16-bit A 16x32 layout: lanes 0-15 hold K={0..7,16..23}, lanes 16-31 K={8..15,24..31}
    const int base0 = g * 8;
    float uf[32];
    #pragma unroll
    for (int seg = 0; seg < 4; ++seg) {
        const v4f* p4 = (const v4f*)(u + seg * 16 + base0);
        v4f x0 = p4[0], x1 = p4[1];
        #pragma unroll
        for (int k = 0; k < 4; ++k) { uf[seg*8 + k] = x0[k]; uf[seg*8 + 4 + k] = x1[k]; }
    }
    v16h a0, a1;
    float usq = 0.0f;
    #pragma unroll
    for (int i = 0; i < 8; ++i) {
        a0[i]     = (_Float16)uf[i];        // K = base0 + i
        a0[8 + i] = (_Float16)uf[8 + i];    // K = 16 + base0 + i
        a1[i]     = (_Float16)uf[16 + i];   // K = 32 + base0 + i
        a1[8 + i] = (_Float16)uf[24 + i];   // K = 48 + base0 + i
    }
    #pragma unroll
    for (int i = 0; i < 32; ++i) usq += uf[i] * uf[i];

    // ---- 5 tiles of 16 item positions; B columns = item vectors ----
    // 16-bit B 32x16 layout: lanes 0-15 hold K={0..7,16..23} (col N=lane), lanes 16-31 K={8..15,24..31}
    const int s = g * 8;
    float zv[5];
    float sq = 0.0f;   // item L2 partial (each lane covers complementary 32 dims of its column)
    #pragma unroll
    for (int t = 0; t < 5; ++t) {
        int p  = t * 16 + n;
        int pc = p > 65 ? 65 : p;                 // clamp padded columns to a valid vector
        int idx = brow[1 + pc];
        const float* iv = embed_item + (size_t)idx * FACTOR;
        v4f y[8];
        #pragma unroll
        for (int j = 0; j < 4; ++j) {
            y[j]     = ((const v4f*)(iv + s))[(j&1) + (j>>1)*4];        // dims s..s+7, 16+s..16+s+7
            y[4 + j] = ((const v4f*)(iv + 32 + s))[(j&1) + (j>>1)*4];   // dims 32+s..32+s+7, 48+s..48+s+7
        }
        v16h b0, b1;
        #pragma unroll
        for (int j = 0; j < 4; ++j)
            #pragma unroll
            for (int k = 0; k < 4; ++k) {
                b0[j*4 + k] = (_Float16)y[j][k];       // K = table order (s = 8*g)
                b1[j*4 + k] = (_Float16)y[4 + j][k];   // K = 32 + table order (s = 8*g)
            }
        if (p < 66) {
            #pragma unroll
            for (int j = 0; j < 8; ++j)
                #pragma unroll
                for (int k = 0; k < 4; ++k) sq += y[j][k] * y[j][k];
        }
        // D = A*B (+C), K=64 via two chained 16x16x32 f16 WMMAs (uniform EXEC here)
        v8f c = {};
        c = __builtin_amdgcn_wmma_f32_16x16x32_f16(false, a0, false, b0, (short)0, c, false, false);
        c = __builtin_amdgcn_wmma_f32_16x16x32_f16(false, a1, false, b1, (short)0, c, false, false);
        zv[t] = c[0];   // all D rows identical (A replicated): c[0] == z[row, p]
    }

    // ---- per-row loss math ----
    float z_ai = __shfl(zv[0], 0, 32);
    float z_aj = __shfl(zv[0], 1, 32);

    float negsum = 0.0f, negmax = -3.402823466e38f;
    if (g == 0) {   // lanes 16-31 hold duplicate z values; count each position once
        #pragma unroll
        for (int t = 0; t < 5; ++t) {
            int p = t * 16 + n;
            if (p >= 2 && p < 66) {
                float zak  = zv[t];
                float term = -log_sigmoid(z_ai - zak) - log_sigmoid(z_aj - zak);
                negsum += term;
                negmax  = fmaxf(negmax, term);
            }
        }
    }
    float l2p = sq + ((lane == 0 || lane == 16) ? usq : 0.0f);

    #pragma unroll
    for (int off = 16; off > 0; off >>= 1) {
        negsum += __shfl_xor(negsum, off, 32);
        l2p    += __shfl_xor(l2p,    off, 32);
        negmax  = fmaxf(negmax, __shfl_xor(negmax, off, 32));
    }

    __shared__ float sh[3][8];
    if (lane == 0) {
        float d        = fabsf(z_ai - z_aj);
        float pos_sim1 = fminf(d, 0.5f);
        float termA    = log_sigmoid(pos_sim1 * 2.0f - negmax);
        sh[0][wid] = termA;   // sum -> loss_posdis
        sh[1][wid] = negsum;  // sum -> loss_m6.mean
        sh[2][wid] = l2p;     // sum -> l2
    }
    __syncthreads();
    if (threadIdx.x == 0) {
        float s0 = 0.f, s1 = 0.f, s2 = 0.f;
        #pragma unroll
        for (int w = 0; w < 8; ++w) { s0 += sh[0][w]; s1 += sh[1][w]; s2 += sh[2][w]; }
        atomicAdd(&ws[0], s0);
        atomicAdd(&ws[1], s1);
        atomicAdd(&ws[2], s2);
    }
}

__global__ void finalize_kernel(const float* __restrict__ ws, float* __restrict__ out) {
    const float invB = 1.0f / (float)BATCH;
    float l2   = 0.01f * ws[2] * invB;
    float loss = (-ws[0] + ws[1]) * invB + l2;
    out[0] = loss;
    out[1] = l2;
}

extern "C" void kernel_launch(void* const* d_in, const int* in_sizes, int n_in,
                              void* d_out, int out_size, void* d_ws, size_t ws_size,
                              hipStream_t stream)
{
    (void)in_sizes; (void)n_in; (void)out_size; (void)ws_size;
    const int*   one_batch  = (const int*)d_in[0];
    const float* embed_user = (const float*)d_in[1];
    const float* embed_item = (const float*)d_in[2];
    float* ws = (float*)d_ws;

    hipMemsetAsync(ws, 0, 3 * sizeof(float), stream);
    bpr_loss_kernel<<<BATCH / 8, 256, 0, stream>>>(one_batch, embed_user, embed_item, ws);
    finalize_kernel<<<1, 1, 0, stream>>>(ws, (float*)d_out);
}
